// MultiHeadAtten_79645873537040
// MI455X (gfx1250) — matrix operand order repaired and checked by reading, not compiled, not run on hardware
//
#include <hip/hip_runtime.h>

// ---------- types ----------
typedef __attribute__((ext_vector_type(16))) _Float16 v16h;
typedef __attribute__((ext_vector_type(8)))  float    v8f;
typedef __attribute__((ext_vector_type(2)))  _Float16 h2;
typedef __attribute__((ext_vector_type(4)))  _Float16 h4;
typedef __attribute__((ext_vector_type(4)))  unsigned int v4u;
typedef __attribute__((ext_vector_type(8)))  int v8i;
typedef __attribute__((ext_vector_type(4)))  int v4i;

union AFrag { v16h v; _Float16 h[16]; h2 p[8]; };

__device__ __forceinline__ v8f wmma_f16(v16h a, v16h b, v8f c) {
  // v_wmma_f32_16x16x32_f16  D = A(16x32) * B(32x16) + C
  return __builtin_amdgcn_wmma_f32_16x16x32_f16(
      /*neg_a=*/false, a, /*neg_b=*/false, b,
      /*c_mod=*/(short)0, c, /*reuse_a=*/false, /*reuse_b=*/false);
}

// A fragment (f16 16x32, row-major src, leading dim ld in halves).
__device__ __forceinline__ v16h load_a_frag(const _Float16* __restrict__ base, int ld) {
  const int lane = threadIdx.x & 31;
  const int m = lane & 15, hf = lane >> 4;
  AFrag f;
#pragma unroll
  for (int v = 0; v < 8; ++v) {
    const int k0 = (v << 1) + (hf << 4);
    f.p[v] = *(const h2*)(base + m * ld + k0);
  }
  return f.v;
}

// B fragment (f16 32x16) from row-major B[k][n] (strided u16 reads).
__device__ __forceinline__ v16h load_b_frag(const _Float16* __restrict__ base, int ld) {
  const int lane = threadIdx.x & 31;
  const int n = lane & 15, hf = lane >> 4;
  AFrag f;
#pragma unroll
  for (int v = 0; v < 8; ++v) {
    const int k0 = (v << 1) + (hf << 4);
    f.h[2 * v]     = base[k0 * ld + n];
    f.h[2 * v + 1] = base[(k0 + 1) * ld + n];
  }
  return f.v;
}

// B fragment for Q*K^T: B[k][n] = Ksrc[n][k] -> contiguous h2 pairs (free transpose).
__device__ __forceinline__ v16h load_bt_frag(const _Float16* __restrict__ kbase, int ld) {
  const int lane = threadIdx.x & 31;
  const int n = lane & 15, hf = lane >> 4;
  AFrag f;
#pragma unroll
  for (int v = 0; v < 8; ++v) {
    const int k0 = (v << 1) + (hf << 4);
    f.p[v] = *(const h2*)(kbase + n * ld + k0);
  }
  return f.v;
}

// ---------- VALU (DPP) 16-lane reductions ----------
template <int CTRL>
__device__ __forceinline__ float dpp_f(float x) {
  return __builtin_bit_cast(float,
      __builtin_amdgcn_mov_dpp(__builtin_bit_cast(int, x), CTRL, 0xf, 0xf, true));
}
__device__ __forceinline__ float red16_max(float x) {
  x = fmaxf(x, dpp_f<0xB1>(x));   // quad_perm [1,0,3,2]  (xor 1)
  x = fmaxf(x, dpp_f<0x4E>(x));   // quad_perm [2,3,0,1]  (xor 2)
  x = fmaxf(x, dpp_f<0x141>(x));  // row_half_mirror
  x = fmaxf(x, dpp_f<0x140>(x));  // row_mirror
  return x;
}
__device__ __forceinline__ float red16_sum(float x) {
  x += dpp_f<0xB1>(x);
  x += dpp_f<0x4E>(x);
  x += dpp_f<0x141>(x);
  x += dpp_f<0x140>(x);
  return x;
}

// ---------- constants ----------
constexpr int Bc = 4, Gc = 64, Nc = 256, Cc = 256, Hc = 8, HDc = 32;
constexpr int MROWS = Bc * Gc * Nc;       // 65536 token rows
constexpr float SM_SCALE = 1.0f / (5.6568542494923806f + 1e-8f); // 1/(sqrt(32)+1e-8)

// ---------- Tensor Data Mover: async 2-D f16 tile (32x32, row stride Cc) -> LDS.
// Descriptor packing per CDNA5 ISA.
__device__ __forceinline__ void tdm_load_tile32x32(const _Float16* gsrc,
                                                   unsigned lds_byte_off) {
  const unsigned long long ga = (unsigned long long)(size_t)gsrc;
  v4u g0;
  g0[0] = 1u;                                      // count=1 (valid), user desc
  g0[1] = lds_byte_off;                            // lds_addr (bytes)
  g0[2] = (unsigned)(ga & 0xffffffffu);            // global_addr[31:0]
  g0[3] = (unsigned)((ga >> 32) & 0x1ffffffu)      // global_addr[56:32]
        | (2u << 30);                              // type=2 ("image")
  v8i g1;
  g1[0] = (int)(1u << 16);       // workgroup_mask=0, data_size=1 (2 bytes)
  g1[1] = (int)(32u << 16);      // tensor_dim0 = 32 (lo16)
  g1[2] = (int)(32u << 16);      // tensor_dim0 hi=0 | tensor_dim1 = 32 (lo16)
  g1[3] = (int)(32u << 16);      // tensor_dim1 hi=0 | tile_dim0 = 32
  g1[4] = 32;                    // tile_dim1 = 32, tile_dim2 = 0
  g1[5] = Cc;                    // tensor_dim0_stride = 256 elements (lo32)
  g1[6] = 0;                     // stride hi | tensor_dim1_stride lo
  g1[7] = 0;
  const v4i gz = {0, 0, 0, 0};   // groups 2/3 unused for 2-D tiles
#if __has_include(<hip/amd_detail/amd_gfx1250_TDM.h>)
  const v8i gz8 = {0, 0, 0, 0, 0, 0, 0, 0};
  __builtin_amdgcn_tensor_load_to_lds(g0, g1, gz, gz, gz8, 0);   // 6-arg toolchain
#else
  __builtin_amdgcn_tensor_load_to_lds(g0, g1, gz, gz, 0);        // 5-arg (ROCm 7.2)
#endif
}

// ======================================================================
// GEMM + bias:  Y(Mx256) = X(Mx256) @ W(256x256) + b
// IN_F32: X f32 -> Y f16 (projections).  else: X f16 -> Y f32 (output proj).
// Block = 256 thr = 8 waves. WG tile 128x64; each wave owns a 32x32 tile
// (2x2 WMMA accumulators -> 4 WMMAs per 32-k chunk, A/B fragments reused 2x).
// ======================================================================
template <bool IN_F32>
__global__ void __launch_bounds__(256)
gemm_bias(const void* __restrict__ Xv, const float* __restrict__ W,
          const float* __restrict__ bias, void* __restrict__ Yv) {
  __shared__ _Float16 Xs[128 * 32];   // 8 KB
  __shared__ _Float16 Ws[32 * 64];    // 4 KB

  const int mblock = blockIdx.x >> 2;
  const int nblock = blockIdx.x & 3;
  const int row0 = mblock * 128;
  const int col0 = nblock * 64;
  const int tid = threadIdx.x;
  const int wave = tid >> 5;
  const int msub = wave >> 1;                    // 0..3 (32-row band)
  const int nsub = wave & 1;                     // 0..1 (32-col band)

  v8f acc00 = {}, acc01 = {}, acc10 = {}, acc11 = {};

#pragma unroll 1
  for (int kc = 0; kc < 8; ++kc) {
    const int k0 = kc * 32;
    __syncthreads();
    {   // X chunk: 128 rows x 32 k, 16 elems/thread (contiguous)
      const int r = tid >> 1;
      const int c = (tid & 1) * 16;
      if constexpr (IN_F32) {
        const float* X = (const float*)Xv;
#pragma unroll
        for (int i = 0; i < 16; ++i)
          Xs[r * 32 + c + i] = (_Float16)X[(row0 + r) * Cc + k0 + c + i];
      } else {
        const _Float16* X = (const _Float16*)Xv;
#pragma unroll
        for (int i = 0; i < 16; ++i)
          Xs[r * 32 + c + i] = X[(row0 + r) * Cc + k0 + c + i];
      }
    }
    {   // W chunk: 32 k x 64 n, 8 elems/thread
      const int r = tid >> 3;
      const int c = (tid & 7) * 8;
#pragma unroll
      for (int i = 0; i < 8; ++i)
        Ws[r * 64 + c + i] = (_Float16)W[(k0 + r) * Cc + col0 + c + i];
    }
    __syncthreads();

    const _Float16* xb = Xs + (msub * 32) * 32;
    const _Float16* wb = Ws + nsub * 32;
    const v16h a0 = load_a_frag(xb, 32);
    const v16h a1 = load_a_frag(xb + 16 * 32, 32);
    const v16h b0 = load_b_frag(wb, 64);
    const v16h b1 = load_b_frag(wb + 16, 64);
    acc00 = wmma_f16(a0, b0, acc00);
    acc01 = wmma_f16(a0, b1, acc01);
    acc10 = wmma_f16(a1, b0, acc10);
    acc11 = wmma_f16(a1, b1, acc11);
  }

  // epilogue: bias + store (D layout: lane holds col n=L%16, rows r+8*(L/16))
  const int lane = tid & 31;
  const int n = lane & 15, hf = lane >> 4;
  const int gc0 = col0 + nsub * 32 + n;
  const int gc1 = gc0 + 16;
  const float bv0 = bias[gc0];
  const float bv1 = bias[gc1];
  const int gr0 = row0 + msub * 32 + 8 * hf;

  if constexpr (IN_F32) {
    _Float16* Y = (_Float16*)Yv;
#pragma unroll
    for (int r = 0; r < 8; ++r) {
      Y[(gr0 + r) * Cc + gc0]      = (_Float16)(acc00[r] + bv0);
      Y[(gr0 + r) * Cc + gc1]      = (_Float16)(acc01[r] + bv1);
      Y[(gr0 + 16 + r) * Cc + gc0] = (_Float16)(acc10[r] + bv0);
      Y[(gr0 + 16 + r) * Cc + gc1] = (_Float16)(acc11[r] + bv1);
    }
  } else {
    float* Y = (float*)Yv;
#pragma unroll
    for (int r = 0; r < 8; ++r) {
      Y[(gr0 + r) * Cc + gc0]      = acc00[r] + bv0;
      Y[(gr0 + r) * Cc + gc1]      = acc01[r] + bv1;
      Y[(gr0 + 16 + r) * Cc + gc0] = acc10[r] + bv0;
      Y[(gr0 + 16 + r) * Cc + gc1] = acc11[r] + bv1;
    }
  }
}

// ======================================================================
// Fused flash attention: one block per (b,g,head). 8 waves x 32 q-rows.
// K/V tiles streamed by the Tensor Data Mover, double-buffered in LDS.
// ======================================================================
__device__ __forceinline__ void softmax_update(v8f& s0, v8f& s1,
                                               float (&rm)[8], float (&rl)[8],
                                               v8f& o0, v8f& o1) {
#pragma unroll
  for (int r = 0; r < 8; ++r) {
    const float a = s0[r] * SM_SCALE;
    const float b = s1[r] * SM_SCALE;
    const float mx = red16_max(fmaxf(a, b));
    const float mnew = fmaxf(rm[r], mx);
    const float corr = __expf(rm[r] - mnew);
    const float e0 = __expf(a - mnew);
    const float e1 = __expf(b - mnew);
    const float rs = red16_sum(e0 + e1);
    rl[r] = rl[r] * corr + rs;
    rm[r] = mnew;
    o0[r] *= corr; o1[r] *= corr;
    s0[r] = e0;   s1[r] = e1;
  }
}

__device__ __forceinline__ void write_p(_Float16* __restrict__ pbuf,
                                        const v8f& s0, const v8f& s1) {
  const int lane = threadIdx.x & 31;
  const int n = lane & 15, hf = lane >> 4;
#pragma unroll
  for (int r = 0; r < 8; ++r) {
    pbuf[(r + 8 * hf) * 32 + n]      = (_Float16)s0[r];
    pbuf[(r + 8 * hf) * 32 + 16 + n] = (_Float16)s1[r];
  }
}

__global__ void __launch_bounds__(256)
attn_kernel(const _Float16* __restrict__ Q, const _Float16* __restrict__ K,
            const _Float16* __restrict__ V, _Float16* __restrict__ CTX) {
  __shared__ _Float16 Kss[2][32 * 32];   // double-buffered TDM targets (4 KB)
  __shared__ _Float16 Vss[2][32 * 32];   // (4 KB)
  __shared__ _Float16 Ps[8][16 * 32];    // per-wave P staging (8 KB)

  const int bg = blockIdx.x >> 3;        // b*G+g
  const int h  = blockIdx.x & 7;
  const int baseRow = bg * Nc;
  const int colOff  = h * HDc;

  const int tid  = threadIdx.x;
  const int wave = tid >> 5;
  const int lane = tid & 31;
  const int n = lane & 15, hf = lane >> 4;
  const int qrow = baseRow + wave * 32;

  const _Float16* Kbase = K + baseRow * Cc + colOff;
  const _Float16* Vbase = V + baseRow * Cc + colOff;

  // Q fragments for this wave's two 16-row tiles (loaded once)
  const v16h aq0 = load_a_frag(Q + qrow * Cc + colOff, Cc);
  const v16h aq1 = load_a_frag(Q + (qrow + 16) * Cc + colOff, Cc);

  v8f o00 = {}, o01 = {}, o10 = {}, o11 = {};
  float rm0[8], rl0[8], rm1[8], rl1[8];
#pragma unroll
  for (int r = 0; r < 8; ++r) { rm0[r] = rm1[r] = -3.0e38f; rl0[r] = rl1[r] = 0.0f; }

  _Float16* pbuf = &Ps[wave][0];

  // kick off DMA of first K/V tile (wave 0 owns the TDM queue + TENSORcnt)
  if (wave == 0) {
    tdm_load_tile32x32(Kbase, (unsigned)(size_t)&Kss[0][0]);
    tdm_load_tile32x32(Vbase, (unsigned)(size_t)&Vss[0][0]);
  }

#pragma unroll 1
  for (int j = 0; j < 8; ++j) {          // stream 32-key blocks
    if (wave == 0) __builtin_amdgcn_s_wait_tensorcnt(0);   // tile j landed
    __syncthreads();                     // publish tile; all prior readers done
    const int cur = j & 1;
    if (wave == 0 && j < 7) {            // overlap next DMA with this block's math
      tdm_load_tile32x32(Kbase + (j + 1) * 32 * Cc, (unsigned)(size_t)&Kss[cur ^ 1][0]);
      tdm_load_tile32x32(Vbase + (j + 1) * 32 * Cc, (unsigned)(size_t)&Vss[cur ^ 1][0]);
    }
    const _Float16* Ksb = &Kss[cur][0];
    const _Float16* Vsb = &Vss[cur][0];

    // S = Q * K^T  (K as B-operand straight from row-major keys)
    const v16h bk0 = load_bt_frag(Ksb, 32);
    const v16h bk1 = load_bt_frag(Ksb + 16 * 32, 32);
    const v8f z = {};
    v8f s00 = wmma_f16(aq0, bk0, z);
    v8f s01 = wmma_f16(aq0, bk1, z);
    v8f s10 = wmma_f16(aq1, bk0, z);
    v8f s11 = wmma_f16(aq1, bk1, z);

    const v16h bv0 = load_b_frag(Vsb, 32);
    const v16h bv1 = load_b_frag(Vsb + 16, 32);

    // ---- m-tile 0 ----
    softmax_update(s00, s01, rm0, rl0, o00, o01);
    write_p(pbuf, s00, s01);
    asm volatile("s_wait_dscnt 0x0" ::: "memory");   // same-wave DS RAW
    {
      const v16h ap = load_a_frag(pbuf, 32);
      o00 = wmma_f16(ap, bv0, o00);
      o01 = wmma_f16(ap, bv1, o01);
    }
    asm volatile("s_wait_dscnt 0x0" ::: "memory");   // reads done before re-write
    // ---- m-tile 1 ----
    softmax_update(s10, s11, rm1, rl1, o10, o11);
    write_p(pbuf, s10, s11);
    asm volatile("s_wait_dscnt 0x0" ::: "memory");
    {
      const v16h ap = load_a_frag(pbuf, 32);
      o10 = wmma_f16(ap, bv0, o10);
      o11 = wmma_f16(ap, bv1, o11);
    }
    asm volatile("s_wait_dscnt 0x0" ::: "memory");
  }

  // normalize + store ctx (f16), layout (B,G,N,C) with C index = h*32 + hd
  _Float16* d0 = CTX + qrow * Cc + colOff;
  _Float16* d1 = CTX + (qrow + 16) * Cc + colOff;
#pragma unroll
  for (int r = 0; r < 8; ++r) {
    const int rr = (r + 8 * hf) * Cc;
    const float i0 = 1.0f / rl0[r];
    const float i1 = 1.0f / rl1[r];
    d0[rr + n]      = (_Float16)(o00[r] * i0);
    d0[rr + 16 + n] = (_Float16)(o01[r] * i0);
    d1[rr + n]      = (_Float16)(o10[r] * i1);
    d1[rr + 16 + n] = (_Float16)(o11[r] * i1);
  }
}

// ======================================================================
extern "C" void kernel_launch(void* const* d_in, const int* in_sizes, int n_in,
                              void* d_out, int out_size, void* d_ws, size_t ws_size,
                              hipStream_t stream) {
  const float* q  = (const float*)d_in[0];
  const float* k  = (const float*)d_in[1];
  const float* v  = (const float*)d_in[2];
  const float* wq = (const float*)d_in[3];
  const float* bq = (const float*)d_in[4];
  const float* wk = (const float*)d_in[5];
  const float* bk = (const float*)d_in[6];
  const float* wv = (const float*)d_in[7];
  const float* bv = (const float*)d_in[8];
  const float* wo = (const float*)d_in[9];
  const float* bo = (const float*)d_in[10];

  _Float16* ws = (_Float16*)d_ws;
  const size_t S = (size_t)MROWS * Cc;          // halves per tensor
  _Float16* Qw = ws;
  _Float16* Kw = ws + S;
  _Float16* Vw = ws + 2 * S;
  _Float16* Cw = ws + 3 * S;

  const dim3 blk(256);
  const int gemm_blocks = (MROWS / 128) * (Cc / 64);   // 512 * 4 = 2048

  gemm_bias<true ><<<gemm_blocks, blk, 0, stream>>>(q, wq, bq, Qw);
  gemm_bias<true ><<<gemm_blocks, blk, 0, stream>>>(k, wk, bk, Kw);
  gemm_bias<true ><<<gemm_blocks, blk, 0, stream>>>(v, wv, bv, Vw);

  attn_kernel<<<Bc * Gc * Hc, blk, 0, stream>>>(Qw, Kw, Vw, Cw);   // 2048 blocks

  gemm_bias<false><<<gemm_blocks, blk, 0, stream>>>(Cw, wo, bo, (float*)d_out);
}
